// GCN_55301998903300
// MI455X (gfx1250) — matrix operand order repaired and checked by reading, not compiled, not run on hardware
//
#include <hip/hip_runtime.h>
#include <stdint.h>

// ---- problem constants (match reference) ----
#define NN   50000      // nodes
#define NE   800000     // edges (no self loops in input)
#define HF   128        // FEAT == HIDDEN
#define NG   128        // graphs
#define REPW 384        // 3*HIDDEN

typedef __attribute__((ext_vector_type(16))) __bf16 v16bf;
typedef __attribute__((ext_vector_type(8)))  float  v8f;

__device__ __forceinline__ unsigned short f32_to_bf16(float f) {
  unsigned int u = __float_as_uint(f);
  u += 0x7FFFu + ((u >> 16) & 1u);      // round-to-nearest-even
  return (unsigned short)(u >> 16);
}

// ---------------- prep kernels ----------------
__global__ void k_deg_init(float* __restrict__ deg) {
  int t = blockIdx.x * blockDim.x + threadIdx.x;
  if (t < NN) deg[t] = 1.0f;            // self-loop contributes 1 to every degree
}

__global__ void k_deg_count(const long long* __restrict__ ei, float* __restrict__ deg) {
  int t = blockIdx.x * blockDim.x + threadIdx.x;
  if (t < NE) atomicAdd(&deg[(int)ei[NE + t]], 1.0f);   // col = ei[1][e]
}

__global__ void k_dis(float* __restrict__ deg) {        // in-place deg -> 1/sqrt(deg)
  int t = blockIdx.x * blockDim.x + threadIdx.x;
  if (t < NN) deg[t] = rsqrtf(deg[t]);                  // deg >= 1 always
}

__global__ void k_edges(const long long* __restrict__ ei, const float* __restrict__ dis,
                        int* __restrict__ row, int* __restrict__ col,
                        float* __restrict__ nrm) {
  int t = blockIdx.x * blockDim.x + threadIdx.x;
  if (t >= NE) return;
  int r = (int)ei[t];
  int c = (int)ei[NE + t];
  row[t] = r; col[t] = c;
  nrm[t] = dis[r] * dis[c];
}

__global__ void k_cvt_x(const float* __restrict__ x, unsigned short* __restrict__ xb) {
  int t = blockIdx.x * blockDim.x + threadIdx.x;
  if (t < NN * HF) xb[t] = f32_to_bf16(x[t]);
}

__global__ void k_cvt_w(const float* __restrict__ W, unsigned short* __restrict__ Wt) {
  // Wt[n][k] = bf16(W[k][n])  (transpose so B loads are K-contiguous)
  int t = blockIdx.x * blockDim.x + threadIdx.x;
  if (t >= HF * HF) return;
  int n = t >> 7, k = t & 127;
  Wt[t] = f32_to_bf16(W[k * HF + n]);
}

__global__ void k_zero(float* __restrict__ p, int n) {
  int t = blockIdx.x * blockDim.x + threadIdx.x;
  if (t < n) p[t] = 0.0f;
}

// ---------------- WMMA GEMM: H = A(bf16 [NN,128]) x W  (Wt = bf16 [n=128][k=128]) ----
// grid = 3125 blocks x 256 threads (8 waves). Wave w computes the 16x16 tile
// (rows m0..m0+15, cols 16w..16w+15). 50000 % 16 == 0 -> EXEC all-ones everywhere.
__global__ void k_gemm_bf16(const unsigned short* __restrict__ A,
                            const unsigned short* __restrict__ Bt,
                            float* __restrict__ H) {
  const int lane = threadIdx.x & 31;
  const int wave = threadIdx.x >> 5;
  const int m0   = blockIdx.x * 16;
  const int n0   = wave * 16;
  const int half = lane >> 4;          // 0: lanes 0-15, 1: lanes 16-31
  const int l16  = lane & 15;

  // A layout (16-bit A 16x32): lane holds row M = lane%16,
  //   vgprs 0-3: K = half*8 .. half*8+7 ; vgprs 4-7: K = 16+half*8 ..
  const size_t arow = (size_t)(m0 + l16) * HF;
  const int    ka   = half * 8;
  // B layout (16-bit B 32x16): lane holds col N = lane%16,
//   vgprs 0-3: K = half*8 .. half*8+7 ; vgprs 4-7: K = 16+half*8 .. (same as A)
  const size_t brow = (size_t)(n0 + l16) * HF;
  const int    kb   = half * 8;

  union AV { v16bf v; uint4 q[2]; };
  v8f c = {};
#pragma unroll
  for (int kk = 0; kk < HF; kk += 32) {
    AV a, b;
    a.q[0] = *(const uint4*)(A + arow + kk + ka);        // 8 bf16, 16B aligned
    a.q[1] = *(const uint4*)(A + arow + kk + 16 + ka);
    b.q[0] = *(const uint4*)(Bt + brow + kk + kb);
    b.q[1] = *(const uint4*)(Bt + brow + kk + 16 + kb);
    c = __builtin_amdgcn_wmma_f32_16x16x32_bf16(false, a.v, false, b.v,
                                                (short)0, c, false, false);
  }

  // C/D layout: vgpr r -> row (half*8 + r), col = n0 + lane%16
  const int rbase = half * 8;
  const int n = n0 + l16;
#pragma unroll
  for (int r = 0; r < 8; ++r)
    H[(size_t)(m0 + rbase + r) * HF + n] = c[r];
}

// ---------------- acc = bias + dis^2 * h   (self-loop folded in) ----------------
__global__ void k_init_acc(const float* __restrict__ h, const float* __restrict__ dis,
                           const float* __restrict__ bias, float* __restrict__ acc) {
  int t = blockIdx.x * blockDim.x + threadIdx.x;
  int node = t >> 5;
  if (node >= NN) return;
  int f = (t & 31) * 4;
  size_t off = (size_t)node * HF + f;
  float d = dis[node], w = d * d;
  float4 hv = *(const float4*)(h + off);
  float4 bv = *(const float4*)(bias + f);
  float4 o;
  o.x = bv.x + w * hv.x; o.y = bv.y + w * hv.y;
  o.z = bv.z + w * hv.z; o.w = bv.w + w * hv.w;
  *(float4*)(acc + off) = o;
}

// ---------------- edge scatter: acc[col] += norm * h[row] ----------------
// one wave32 per edge; lane handles a float4 slice (coalesced 512B per edge)
__global__ void k_scatter(const int* __restrict__ row, const int* __restrict__ col,
                          const float* __restrict__ nrm, const float* __restrict__ h,
                          float* __restrict__ acc) {
  int e = (blockIdx.x * blockDim.x + threadIdx.x) >> 5;
  if (e >= NE) return;
  int lane = threadIdx.x & 31;
  int r = row[e], c = col[e];
  float w = nrm[e];
  float4 v = *(const float4*)(h + (size_t)r * HF + lane * 4);
  float* op = acc + (size_t)c * HF + lane * 4;
  atomicAdd(op + 0, w * v.x);
  atomicAdd(op + 1, w * v.y);
  atomicAdd(op + 2, w * v.z);
  atomicAdd(op + 3, w * v.w);
}

// ---------------- ReLU + (next-layer bf16) + (x3 out) + graph pooling ----------------
__global__ void k_finalize(const float* __restrict__ acc, const long long* __restrict__ batch,
                           unsigned short* __restrict__ xb_next,   // null on last layer
                           float* __restrict__ x3,                 // non-null on last layer
                           float* __restrict__ rep, int layer) {
  int t = blockIdx.x * blockDim.x + threadIdx.x;
  int node = t >> 5;
  if (node >= NN) return;
  int f = (t & 31) * 4;
  size_t off = (size_t)node * HF + f;
  float4 v = *(const float4*)(acc + off);
  v.x = fmaxf(v.x, 0.0f); v.y = fmaxf(v.y, 0.0f);
  v.z = fmaxf(v.z, 0.0f); v.w = fmaxf(v.w, 0.0f);
  if (xb_next) {
    unsigned short* q = xb_next + off;
    q[0] = f32_to_bf16(v.x); q[1] = f32_to_bf16(v.y);
    q[2] = f32_to_bf16(v.z); q[3] = f32_to_bf16(v.w);
  }
  if (x3) *(float4*)(x3 + off) = v;
  int g = (int)batch[node];
  float* rp = rep + (size_t)g * REPW + layer * HF + f;
  atomicAdd(rp + 0, v.x);
  atomicAdd(rp + 1, v.y);
  atomicAdd(rp + 2, v.z);
  atomicAdd(rp + 3, v.w);
}

// ---------------- host ----------------
extern "C" void kernel_launch(void* const* d_in, const int* in_sizes, int n_in,
                              void* d_out, int out_size, void* d_ws, size_t ws_size,
                              hipStream_t stream) {
  const float*     x     = (const float*)d_in[0];
  const long long* ei    = (const long long*)d_in[1];   // [2, NE] int64
  const long long* batch = (const long long*)d_in[2];   // [NN] int64
  const float* W[3] = { (const float*)d_in[3], (const float*)d_in[5], (const float*)d_in[7] };
  const float* B[3] = { (const float*)d_in[4], (const float*)d_in[6], (const float*)d_in[8] };

  float* rep = (float*)d_out;              // [NG, 384]
  float* x3  = rep + NG * REPW;            // [NN, 128]

  // workspace carve-out (~74 MB)
  char* p = (char*)d_ws;
  auto carve = [&](size_t bytes) { void* r = p; p += (bytes + 255) & ~(size_t)255; return r; };
  float*          dis  = (float*)carve((size_t)NN * 4);          // deg -> dis in place
  int*            row  = (int*)carve((size_t)NE * 4);
  int*            col  = (int*)carve((size_t)NE * 4);
  float*          nrm  = (float*)carve((size_t)NE * 4);
  unsigned short* xb   = (unsigned short*)carve((size_t)NN * HF * 2);
  float*          h    = (float*)carve((size_t)NN * HF * 4);
  float*          acc  = (float*)carve((size_t)NN * HF * 4);
  unsigned short* Wt[3];
  for (int l = 0; l < 3; ++l) Wt[l] = (unsigned short*)carve((size_t)HF * HF * 2);

  const int TB = 256;
  const int gN    = (NN + TB - 1) / TB;          // per-node
  const int gE    = (NE + TB - 1) / TB;          // per-edge
  const int gNF   = (NN * HF + TB - 1) / TB;     // per-feature element
  const int gN32  = (NN * 32 + TB - 1) / TB;     // node x float4-lane
  const int gE32  = (NE * 32 + TB - 1) / TB;     // edge-wave
  const int gW    = (HF * HF + TB - 1) / TB;
  const int gRep  = (NG * REPW + TB - 1) / TB;
  const int gGemm = NN / 16;                     // 3125, exact

  // normalization
  k_deg_init<<<gN, TB, 0, stream>>>(dis);
  k_deg_count<<<gE, TB, 0, stream>>>(ei, dis);
  k_dis<<<gN, TB, 0, stream>>>(dis);
  k_edges<<<gE, TB, 0, stream>>>(ei, dis, row, col, nrm);

  // precision prep
  k_cvt_x<<<gNF, TB, 0, stream>>>(x, xb);
  for (int l = 0; l < 3; ++l) k_cvt_w<<<gW, TB, 0, stream>>>(W[l], Wt[l]);
  k_zero<<<gRep, TB, 0, stream>>>(rep, NG * REPW);

  // three GCN layers
  for (int l = 0; l < 3; ++l) {
    k_gemm_bf16<<<gGemm, TB, 0, stream>>>(xb, Wt[l], h);
    k_init_acc<<<gN32, TB, 0, stream>>>(h, dis, B[l], acc);
    k_scatter<<<gE32, TB, 0, stream>>>(row, col, nrm, h, acc);
    k_finalize<<<gN32, TB, 0, stream>>>(acc, batch,
                                        (l < 2) ? xb : (unsigned short*)nullptr,
                                        (l == 2) ? x3 : (float*)nullptr,
                                        rep, l);
  }
}
